// attention_59124519797335
// MI455X (gfx1250) — matrix operand order repaired and checked by reading, not compiled, not run on hardware
//
#include <hip/hip_runtime.h>

#define SEQ   8192
#define DIMS  1024
#define NHEAD 16
#define HD    64
#define LAYERN 4

typedef _Float16 f16;
typedef __attribute__((ext_vector_type(16))) _Float16 v16h;
typedef __attribute__((ext_vector_type(8)))  float    v8f;
typedef __attribute__((ext_vector_type(4)))  int      v4i;

union AFrag { v16h h; unsigned int u[8]; };

#define GLOBAL_AS __attribute__((address_space(1)))
#define LDS_AS    __attribute__((address_space(3)))

#if defined(__has_builtin)
#if __has_builtin(__builtin_amdgcn_global_load_async_to_lds_b128) && \
    __has_builtin(__builtin_amdgcn_s_wait_asynccnt)
#define ASYNC_LDS 1
#endif
#endif

// 16-byte global -> LDS copy: async-to-LDS (ASYNCcnt) if available, else via VGPRs.
__device__ __forceinline__ void cp16_g2s(const f16* g, f16* s) {
#ifdef ASYNC_LDS
  __builtin_amdgcn_global_load_async_to_lds_b128(
      (GLOBAL_AS v4i*)g, (LDS_AS v4i*)s, 0, 0);
#else
  *(uint4*)s = *(const uint4*)g;
#endif
}
__device__ __forceinline__ void cp_wait() {
#ifdef ASYNC_LDS
  __builtin_amdgcn_s_wait_asynccnt(0);
#endif
}

// ---------------------------------------------------------------------------
// 1) LayerNorm prep: one block per row; mean/var shared by q/kv paths, mag=mean.
// ---------------------------------------------------------------------------
__global__ __launch_bounds__(256) void ln_prep_kernel(
    const float* __restrict__ x,
    const float* __restrict__ qg, const float* __restrict__ qb,
    const float* __restrict__ kg, const float* __restrict__ kb,
    f16* __restrict__ xq, f16* __restrict__ xkv, float* __restrict__ mag) {
  __shared__ float red[256];
  const int row = blockIdx.x;
  const float* xr = x + (size_t)row * DIMS;
  float vals[4], lsum = 0.f;
  #pragma unroll
  for (int i = 0; i < 4; ++i) { vals[i] = xr[threadIdx.x + 256 * i]; lsum += vals[i]; }
  red[threadIdx.x] = lsum; __syncthreads();
  for (int s = 128; s > 0; s >>= 1) { if (threadIdx.x < s) red[threadIdx.x] += red[threadIdx.x + s]; __syncthreads(); }
  const float mean = red[0] * (1.0f / DIMS);
  __syncthreads();
  float lv = 0.f;
  #pragma unroll
  for (int i = 0; i < 4; ++i) { float d = vals[i] - mean; lv += d * d; }
  red[threadIdx.x] = lv; __syncthreads();
  for (int s = 128; s > 0; s >>= 1) { if (threadIdx.x < s) red[threadIdx.x] += red[threadIdx.x + s]; __syncthreads(); }
  const float rstd = rsqrtf(red[0] * (1.0f / DIMS) + 1e-5f);
  if (threadIdx.x == 0) mag[row] = mean;
  #pragma unroll
  for (int i = 0; i < 4; ++i) {
    int c = threadIdx.x + 256 * i;
    float n = (vals[i] - mean) * rstd;
    xq [(size_t)row * DIMS + c] = (f16)(n * qg[c] + qb[c]);
    xkv[(size_t)row * DIMS + c] = (f16)(n * kg[c] + kb[c]);
  }
}

// ---------------------------------------------------------------------------
// 2) Weight convert + transpose: Wt[n*K+k] = (f16)W[k*N+n]
// ---------------------------------------------------------------------------
__global__ __launch_bounds__(256) void wconv_kernel(
    const float* __restrict__ W, f16* __restrict__ Wt, int K, int N) {
  __shared__ float tile[32][33];
  const int kbt = blockIdx.x * 32, nb = blockIdx.y * 32;
  const int tx = threadIdx.x & 31, ty = threadIdx.x >> 5;
  #pragma unroll
  for (int i = 0; i < 32; i += 8)
    tile[ty + i][tx] = W[(size_t)(kbt + ty + i) * N + nb + tx];
  __syncthreads();
  #pragma unroll
  for (int i = 0; i < 32; i += 8)
    Wt[(size_t)(nb + ty + i) * K + kbt + tx] = (f16)tile[tx][ty + i];
}

// ---------------------------------------------------------------------------
// 3) WMMA GEMM: C = A[M,K]f16 * Bt[N,K]^T + bias (f32 out).
//    Workgroup 256Mx64N (8 waves x 32Mx64N). B K-step tile staged in LDS
//    (shared by all 8 waves, async-to-LDS when available).
// ---------------------------------------------------------------------------
__global__ __launch_bounds__(256) void gemm_wmma_kernel(
    const f16* __restrict__ A, const f16* __restrict__ Bt,
    const float* __restrict__ bias, float* __restrict__ C,
    int M, int N, int K) {
  __shared__ f16 ldsB[64 * 32];          // [n][k] 4KB
  const int wave = threadIdx.x >> 5, lane = threadIdx.x & 31;
  const int m0 = blockIdx.x * 256 + wave * 32;
  const int n0 = blockIdx.y * 64;
  const int alo = (lane < 16) ? 0 : 8;
  const int blo = (lane < 16) ? 0 : 16;
  v8f acc[2][4] = {};
  const f16* arow0 = A + (size_t)(m0 +      (lane & 15)) * K;
  const f16* arow1 = A + (size_t)(m0 + 16 + (lane & 15)) * K;
  // staging indices for this thread (64 rows x 32 k, 16B per thread)
  const int sn = threadIdx.x >> 2, sk = (threadIdx.x & 3) * 8;
  const f16* bsrc = Bt + (size_t)(n0 + sn) * K + sk;
  f16* bdst = ldsB + sn * 32 + sk;

  for (int k0 = 0; k0 < K; k0 += 32) {
    __syncthreads();                     // protect ldsB from prior-iter readers
    cp16_g2s(bsrc + k0, bdst);
    cp_wait();
    __syncthreads();
    AFrag a0, a1;
    #pragma unroll
    for (int e = 0; e < 8; ++e) {
      int koff = k0 + ((e < 4) ? 2 * e : 16 + 2 * (e - 4)) + alo;
      a0.u[e] = *(const unsigned int*)(arow0 + koff);
      a1.u[e] = *(const unsigned int*)(arow1 + koff);
    }
    #pragma unroll
    for (int c = 0; c < 4; ++c) {
      const f16* brow = ldsB + (c * 16 + (lane & 15)) * 32;
      AFrag b;
      #pragma unroll
      for (int e = 0; e < 8; ++e)
        b.u[e] = *(const unsigned int*)(brow + ((e < 4) ? 2 * e : 16 + 2 * (e - 4)) + alo);
      acc[0][c] = __builtin_amdgcn_wmma_f32_16x16x32_f16(
          false, a0.h, false, b.h, (short)0, acc[0][c], false, false);
      acc[1][c] = __builtin_amdgcn_wmma_f32_16x16x32_f16(
          false, a1.h, false, b.h, (short)0, acc[1][c], false, false);
    }
  }
  #pragma unroll
  for (int c = 0; c < 4; ++c) {
    int col = n0 + c * 16 + (lane & 15);
    float bv = bias[col];
    #pragma unroll
    for (int sub = 0; sub < 2; ++sub)
      #pragma unroll
      for (int r = 0; r < 8; ++r)
        C[(size_t)(m0 + sub * 16 + alo + r) * N + col] = acc[sub][c][r] + bv;
  }
}

// ---------------------------------------------------------------------------
// 4) RoPE + per-head LN. One wave per (index j, head); 32 lanes = 32 pairs.
// ---------------------------------------------------------------------------
__global__ __launch_bounds__(256) void rope_ln_kernel(
    const float* __restrict__ src, int srcN,
    const float* __restrict__ mag,
    const float* __restrict__ g, const float* __restrict__ b,
    const int* __restrict__ skip,
    f16* __restrict__ out, int is_k) {
  const int wid  = blockIdx.x * 8 + (threadIdx.x >> 5);
  const int lane = threadIdx.x & 31;
  const int j = wid >> 4, h = wid & 15;
  int stride = 1;
  if (is_k) { stride = LAYERN - skip[0]; if (stride < 1) stride = 1; }
  const long pos = (long)j * stride;
  f16* op = out + ((size_t)h * SEQ + j) * HD + 2 * lane;
  if (pos >= SEQ) { op[0] = (f16)0.f; op[1] = (f16)0.f; return; }
  const int d0 = 2 * lane;
  const float xr = src[pos * srcN + (size_t)h * HD + d0];
  const float xi = src[pos * srcN + (size_t)h * HD + d0 + 1];
  const float LN21 = 3.04452243772342300f;
  const float fr = 0.2f * (expf(LN21 * ((float)lane * (1.0f / 31.0f))) - 1.0f);
  const float ang = (float)pos * fr;
  const float ca = cosf(ang), sa = sinf(ang);
  const float m = mag[pos];
  float outr = m * (xr * ca - xi * sa);
  float outi = m * (xr * sa + xi * ca);
  float s1 = outr + outi;
  float s2 = outr * outr + outi * outi;
  for (int mk = 1; mk < 32; mk <<= 1) {
    s1 += __shfl_xor(s1, mk, 32);
    s2 += __shfl_xor(s2, mk, 32);
  }
  const float mean = s1 * (1.0f / HD);
  const float var  = s2 * (1.0f / HD) - mean * mean;
  const float rstd = rsqrtf(var + 1e-5f);
  op[0] = (f16)((outr - mean) * rstd * g[d0]     + b[d0]);
  op[1] = (f16)((outi - mean) * rstd * g[d0 + 1] + b[d0 + 1]);
}

// ---------------------------------------------------------------------------
// 5) V select + transpose: vT[h][d][j] = kv_raw[j*stride][1024 + h*64 + d]
// ---------------------------------------------------------------------------
__global__ __launch_bounds__(256) void vsel_kernel(
    const float* __restrict__ kv, const int* __restrict__ skip,
    f16* __restrict__ vT) {
  int stride = LAYERN - skip[0]; if (stride < 1) stride = 1;
  const size_t idx = (size_t)blockIdx.x * 256 + threadIdx.x;
  const int j = (int)(idx & (SEQ - 1));
  const int d = (int)((idx >> 13) & (HD - 1));
  const int h = (int)(idx >> 19);
  const long pos = (long)j * stride;
  float v = (pos < SEQ) ? kv[pos * (2 * DIMS) + DIMS + (size_t)h * HD + d] : 0.0f;
  vT[idx] = (f16)v;
}

// ---------------------------------------------------------------------------
// 6) Flash attention. Workgroup = 8 waves x 16 queries, SAME head.
//    Per 32-key block: K/V tiles staged once in LDS (async when available),
//    shared by all 8 waves -> 8x global traffic cut. 4 WMMA QK^T + f32 online
//    softmax (half-wave shfl) + P restripe via LDS + 4 WMMA P*V.
// ---------------------------------------------------------------------------
__global__ __launch_bounds__(256) void attn_kernel(
    const f16* __restrict__ qn, const f16* __restrict__ kn,
    const f16* __restrict__ vT, const int* __restrict__ skip,
    f16* __restrict__ aout) {
  __shared__ f16 kt[32 * HD];       // [key][dim]  4KB
  __shared__ f16 vt[HD * 32];       // [dim][key]  4KB
  __shared__ f16 pls[8][16 * 32];   // per-wave P  8KB
  const int wave = threadIdx.x >> 5, lane = threadIdx.x & 31;
  const int h  = blockIdx.x >> 6;                 // 64 q-groups per head
  const int q0 = ((blockIdx.x & 63) * 8 + wave) * 16;
  int stride = LAYERN - skip[0]; if (stride < 1) stride = 1;
  const int kvlen = (SEQ + stride - 1) / stride;
  const int alo = (lane < 16) ? 0 : 8;
  const int blo = (lane < 16) ? 0 : 16;

  // Q fragments (M=16, K=64 -> two 16x32 A-frags), loaded once
  const f16* qrow = qn + ((size_t)h * SEQ + q0 + (lane & 15)) * HD;
  AFrag qa[2];
  #pragma unroll
  for (int f = 0; f < 2; ++f)
    #pragma unroll
    for (int e = 0; e < 8; ++e) {
      int koff = f * 32 + ((e < 4) ? 2 * e : 16 + 2 * (e - 4)) + alo;
      qa[f].u[e] = *(const unsigned int*)(qrow + koff);
    }

  // staging indices (16B per thread per tile)
  const int skey = threadIdx.x >> 3, skd = (threadIdx.x & 7) * 8;   // kt
  const int svd = threadIdx.x >> 2, svk = (threadIdx.x & 3) * 8;    // vt
  const f16* ksrc = kn + ((size_t)h * SEQ + skey) * HD + skd;
  const f16* vsrc = vT + ((size_t)h * HD + svd) * SEQ + svk;
  f16* kdst = kt + skey * HD + skd;
  f16* vdst = vt + svd * 32 + svk;

  v8f oacc[4] = {};
  float mrow[8], lrow[8];
  #pragma unroll
  for (int r = 0; r < 8; ++r) { mrow[r] = -1e30f; lrow[r] = 0.f; }
  const float scale = 0.125f;  // 1/sqrt(64)
  const int nblk = (kvlen + 31) >> 5;
  f16* pbuf = &pls[wave][0];

  for (int kbk = 0; kbk < nblk; ++kbk) {
    const int kbase = kbk * 32;
    __syncthreads();                       // protect kt/vt from prior readers
    cp16_g2s(ksrc + (size_t)kbase * HD, kdst);
    cp16_g2s(vsrc + kbase, vdst);
    cp_wait();
    __syncthreads();
    // S = Q * K^T (two 16-key chunks, K=64 split in 2 WMMA each)
    v8f s[2];
    #pragma unroll
    for (int c = 0; c < 2; ++c) {
      v8f sc = {};
      const f16* krow = kt + (c * 16 + (lane & 15)) * HD;
      #pragma unroll
      for (int f = 0; f < 2; ++f) {
        AFrag kb;
        #pragma unroll
        for (int e = 0; e < 8; ++e)
          kb.u[e] = *(const unsigned int*)(krow + f * 32 + ((e < 4) ? 2 * e : 16 + 2 * (e - 4)) + alo);
        sc = __builtin_amdgcn_wmma_f32_16x16x32_f16(
            false, qa[f].h, false, kb.h, (short)0, sc, false, false);
      }
      s[c] = sc;
    }
    // scale + mask keys >= kvlen
    #pragma unroll
    for (int c = 0; c < 2; ++c) {
      const float msk = ((kbase + c * 16 + (lane & 15)) < kvlen) ? 0.0f : -1e30f;
      #pragma unroll
      for (int r = 0; r < 8; ++r) s[c][r] = s[c][r] * scale + msk;
    }
    // online softmax: rows 0..7 in lanes 0..15, rows 8..15 in lanes 16..31
    float alpha[8];
    #pragma unroll
    for (int r = 0; r < 8; ++r) {
      float mx = fmaxf(s[0][r], s[1][r]);
      for (int mk = 1; mk < 16; mk <<= 1) mx = fmaxf(mx, __shfl_xor(mx, mk, 32));
      float mnew = fmaxf(mrow[r], mx);
      alpha[r] = expf(mrow[r] - mnew);
      mrow[r] = mnew;
      float p0 = expf(s[0][r] - mnew);
      float p1 = expf(s[1][r] - mnew);
      s[0][r] = p0; s[1][r] = p1;
      float ps = p0 + p1;
      for (int mk = 1; mk < 16; mk <<= 1) ps += __shfl_xor(ps, mk, 32);
      lrow[r] = lrow[r] * alpha[r] + ps;
    }
    #pragma unroll
    for (int c = 0; c < 4; ++c)
      #pragma unroll
      for (int r = 0; r < 8; ++r) oacc[c][r] *= alpha[r];
    // P (C-layout) -> per-wave LDS -> A-fragment layout (in-wave, in-order DS)
    #pragma unroll
    for (int c = 0; c < 2; ++c)
      #pragma unroll
      for (int r = 0; r < 8; ++r)
        pbuf[(alo + r) * 32 + c * 16 + (lane & 15)] = (f16)s[c][r];
    AFrag pa;
    #pragma unroll
    for (int e = 0; e < 8; ++e) {
      int koff = ((e < 4) ? 2 * e : 16 + 2 * (e - 4)) + alo;
      pa.u[e] = *(const unsigned int*)(pbuf + (lane & 15) * 32 + koff);
    }
    // O += P * V
    #pragma unroll
    for (int c = 0; c < 4; ++c) {
      const f16* vrow = vt + (c * 16 + (lane & 15)) * 32;
      AFrag vb;
      #pragma unroll
      for (int e = 0; e < 8; ++e)
        vb.u[e] = *(const unsigned int*)(vrow + ((e < 4) ? 2 * e : 16 + 2 * (e - 4)) + alo);
      oacc[c] = __builtin_amdgcn_wmma_f32_16x16x32_f16(
          false, pa.h, false, vb.h, (short)0, oacc[c], false, false);
    }
  }
  #pragma unroll
  for (int c = 0; c < 4; ++c) {
    int col = h * HD + c * 16 + (lane & 15);
    #pragma unroll
    for (int r = 0; r < 8; ++r)
      aout[(size_t)(q0 + alo + r) * DIMS + col] = (f16)(oacc[c][r] / lrow[r]);
  }
}

// ---------------------------------------------------------------------------
extern "C" void kernel_launch(void* const* d_in, const int* in_sizes, int n_in,
                              void* d_out, int out_size, void* d_ws, size_t ws_size,
                              hipStream_t stream) {
  const float* x     = (const float*)d_in[0];
  const float* qn_g  = (const float*)d_in[1];
  const float* qn_b  = (const float*)d_in[2];
  const float* Wq    = (const float*)d_in[3];
  const float* bq    = (const float*)d_in[4];
  const float* kvn_g = (const float*)d_in[5];
  const float* kvn_b = (const float*)d_in[6];
  const float* Wkv   = (const float*)d_in[7];
  const float* bkv   = (const float*)d_in[8];
  const float* Wout  = (const float*)d_in[9];
  const float* bout  = (const float*)d_in[10];
  const float* ln_g  = (const float*)d_in[11];
  const float* ln_b  = (const float*)d_in[12];
  const int*   skip  = (const int*)d_in[13];
  float* out = (float*)d_out;

  char* ws = (char*)d_ws;
  size_t off = 0;
  auto alloc = [&](size_t bytes) -> void* {
    void* p = ws + off; off += (bytes + 255) & ~(size_t)255; return p;
  };
  f16*   xq_h    = (f16*)  alloc((size_t)SEQ * DIMS * 2);
  f16*   xkv_h   = (f16*)  alloc((size_t)SEQ * DIMS * 2);
  float* mag     = (float*)alloc((size_t)SEQ * 4);
  f16*   WqT     = (f16*)  alloc((size_t)DIMS * DIMS * 2);
  f16*   WkvT    = (f16*)  alloc((size_t)2 * DIMS * DIMS * 2);
  f16*   WoutT   = (f16*)  alloc((size_t)DIMS * DIMS * 2);
  float* q_raw   = (float*)alloc((size_t)SEQ * DIMS * 4);
  float* kv_raw  = (float*)alloc((size_t)SEQ * 2 * DIMS * 4);
  f16*   qn_h    = (f16*)  alloc((size_t)NHEAD * SEQ * HD * 2);
  f16*   kn_h    = (f16*)  alloc((size_t)NHEAD * SEQ * HD * 2);
  f16*   vT_h    = (f16*)  alloc((size_t)NHEAD * HD * SEQ * 2);
  f16*   a_h     = (f16*)  alloc((size_t)SEQ * DIMS * 2);
  (void)ws_size; (void)in_sizes; (void)n_in; (void)out_size;

  ln_prep_kernel<<<SEQ, 256, 0, stream>>>(x, qn_g, qn_b, kvn_g, kvn_b, xq_h, xkv_h, mag);
  wconv_kernel<<<dim3(DIMS / 32, DIMS / 32), 256, 0, stream>>>(Wq, WqT, DIMS, DIMS);
  wconv_kernel<<<dim3(DIMS / 32, 2 * DIMS / 32), 256, 0, stream>>>(Wkv, WkvT, DIMS, 2 * DIMS);
  wconv_kernel<<<dim3(DIMS / 32, DIMS / 32), 256, 0, stream>>>(Wout, WoutT, DIMS, DIMS);
  gemm_wmma_kernel<<<dim3(SEQ / 256, DIMS / 64), 256, 0, stream>>>(
      xq_h, WqT, bq, q_raw, SEQ, DIMS, DIMS);
  gemm_wmma_kernel<<<dim3(SEQ / 256, 2 * DIMS / 64), 256, 0, stream>>>(
      xkv_h, WkvT, bkv, kv_raw, SEQ, 2 * DIMS, DIMS);
  const int rope_blocks = SEQ * NHEAD / 8;
  rope_ln_kernel<<<rope_blocks, 256, 0, stream>>>(q_raw, DIMS, mag, ln_g, ln_b, skip, qn_h, 0);
  rope_ln_kernel<<<rope_blocks, 256, 0, stream>>>(kv_raw, 2 * DIMS, mag, ln_g, ln_b, skip, kn_h, 1);
  vsel_kernel<<<(NHEAD * HD * SEQ) / 256, 256, 0, stream>>>(kv_raw, skip, vT_h);
  attn_kernel<<<NHEAD * (SEQ / 128), 256, 0, stream>>>(qn_h, kn_h, vT_h, skip, a_h);
  gemm_wmma_kernel<<<dim3(SEQ / 256, DIMS / 64), 256, 0, stream>>>(
      a_h, WoutT, bout, out, SEQ, DIMS, DIMS);
}
